// LlamaSparseAttention_12266426597392
// MI455X (gfx1250) — matrix operand order repaired and checked by reading, not compiled, not run on hardware
//
#include <hip/hip_runtime.h>
#include <hip/hip_bf16.h>
#include <math.h>

typedef __attribute__((ext_vector_type(16))) __bf16 v16bf;
typedef __attribute__((ext_vector_type(8)))  __bf16 v8bf;
typedef __attribute__((ext_vector_type(8)))  float  v8f;

#define HIDDEN 2048
#define QLEN   2048
#define NHEADS 32
#define NKV    8
#define HD     64
#define KVDIM  (NKV*HD)   // 512

__device__ __forceinline__ __bf16 f2bf(float f) {
  unsigned u = __builtin_bit_cast(unsigned, f);
  unsigned r = u + 0x7FFFu + ((u >> 16) & 1u);
  unsigned short h = (unsigned short)(r >> 16);
  return __builtin_bit_cast(__bf16, h);
}

// ---- CDNA5 async global->LDS copy (bypasses VGPRs, tracked by ASYNCcnt) ----
__device__ __forceinline__ unsigned lds_off(const void* p) {
  return (unsigned)(unsigned long long)(__attribute__((address_space(3))) const void*)p;
}
__device__ __forceinline__ void async_b128(unsigned lds, const void* g) {
  asm volatile("global_load_async_to_lds_b128 %0, %1, off"
               :: "v"(lds), "v"((unsigned long long)g) : "memory");
}
template <int N> __device__ __forceinline__ void wait_async() {
  asm volatile("s_wait_asynccnt %0" :: "n"(N) : "memory");
}

// ---------------- fp32 -> bf16 conversion ----------------
__global__ void cvt_bf16_kernel(const float* __restrict__ src, __bf16* __restrict__ dst, int n) {
  int i = blockIdx.x * blockDim.x + threadIdx.x;
  if (i < n) dst[i] = f2bf(src[i]);
}

// ---------------- head predictor + top-16 mask ----------------
__global__ __launch_bounds__(256) void headmask_kernel(const float* __restrict__ X,
                                                       const float* __restrict__ Wp,
                                                       float* __restrict__ maskbuf) {
  int w = threadIdx.x >> 5, lane = threadIdx.x & 31;
  int q = blockIdx.x * 8 + w;
  const float* xr = X + (size_t)q * HIDDEN;
  const float* wr = Wp + (size_t)lane * HIDDEN;
  float acc = 0.f;
  for (int k = 0; k < HIDDEN; ++k) acc += xr[k] * wr[k];
  int cnt = 0;
  for (int j = 0; j < 32; ++j) {
    float o = __shfl(acc, j, 32);
    cnt += (o > acc) || (o == acc && j < lane);
  }
  maskbuf[(size_t)lane * QLEN + q] = (cnt < 16) ? 1.f : 0.f;
}

// ---------------- bf16 WMMA GEMM: C[M,N] = A[M,K] @ B[N,K]^T ----------------
// block 256 = 8 waves; tile 128x128x32; wave = 64x32 (4x2 wmma tiles)
// double-buffered LDS, async global->LDS staging overlapped with WMMA
__global__ __launch_bounds__(256) void gemm_bf16_kernel(const __bf16* __restrict__ A,
                                                        const __bf16* __restrict__ B,
                                                        float* __restrict__ C,
                                                        int M, int N, int K) {
  __shared__ __align__(16) __bf16 As[2][128 * 40];
  __shared__ __align__(16) __bf16 Bs[2][128 * 40];
  int t = threadIdx.x, lane = t & 31, w = t >> 5;
  int mbase = blockIdx.y * 128, nbase = blockIdx.x * 128;
  int wm = (w >> 2) * 64, wn = (w & 3) * 32;
  int ml = lane & 15, h = lane >> 4;

  v8f acc[4][2];
  for (int mt = 0; mt < 4; ++mt)
    for (int nt = 0; nt < 2; ++nt)
      acc[mt][nt] = v8f{0, 0, 0, 0, 0, 0, 0, 0};

  auto stage = [&](int p, int kb) {
#pragma unroll
    for (int r = 0; r < 2; ++r) {
      int c = r * 256 + t;
      int row = c >> 2, col = (c & 3) * 8;
      async_b128(lds_off(&As[p][row * 40 + col]), &A[(size_t)(mbase + row) * K + kb + col]);
      async_b128(lds_off(&Bs[p][row * 40 + col]), &B[(size_t)(nbase + row) * K + kb + col]);
    }
  };

  int nk = K >> 5;
  stage(0, 0);
  for (int i = 0; i < nk; ++i) {
    int p = i & 1;
    if (i + 1 < nk) {
      stage(p ^ 1, (i + 1) << 5);
      wait_async<4>();      // 4 next-stage ops in flight -> current stage landed
    } else {
      wait_async<0>();
    }
    __syncthreads();

    v16bf af[4], bfr[2];
#pragma unroll
    for (int mt = 0; mt < 4; ++mt) {
      const __bf16* pA = &As[p][(wm + mt * 16 + ml) * 40 + h * 8];
      v8bf lo = *(const v8bf*)pA, hi = *(const v8bf*)(pA + 16);
#pragma unroll
      for (int e = 0; e < 8; ++e) { af[mt][e] = lo[e]; af[mt][e + 8] = hi[e]; }
    }
#pragma unroll
    for (int nt = 0; nt < 2; ++nt) {
      const __bf16* pB = &Bs[p][(wn + nt * 16 + ml) * 40 + h * 8];
      v8bf lo = *(const v8bf*)pB, hi = *(const v8bf*)(pB + 16);
#pragma unroll
      for (int e = 0; e < 8; ++e) { bfr[nt][e] = lo[e]; bfr[nt][e + 8] = hi[e]; }
    }
#pragma unroll
    for (int mt = 0; mt < 4; ++mt)
#pragma unroll
      for (int nt = 0; nt < 2; ++nt)
        acc[mt][nt] = __builtin_amdgcn_wmma_f32_16x16x32_bf16(
            false, af[mt], false, bfr[nt], (short)0, acc[mt][nt], false, false);
    __syncthreads();   // everyone done reading buf p; it may be refilled next iter
  }

#pragma unroll
  for (int mt = 0; mt < 4; ++mt)
#pragma unroll
    for (int nt = 0; nt < 2; ++nt)
#pragma unroll
      for (int r = 0; r < 8; ++r) {
        int row = mbase + wm + mt * 16 + r + (h << 3);
        int col = nbase + wn + nt * 16 + ml;
        C[(size_t)row * N + col] = acc[mt][nt][r];
      }
}

// ---------------- RoPE + repack ----------------
__global__ void rope_q_kernel(const float* __restrict__ Qf, const int* __restrict__ pid,
                              __bf16* __restrict__ Qb) {
  int t = blockIdx.x * blockDim.x + threadIdx.x; // QLEN*32*32
  int i = t & 31, head = (t >> 5) & 31, q = t >> 10;
  float pos = (float)pid[q];
  float inv = __expf(-(float)(2 * i) / (float)HD * __logf(10000.0f));
  float s, c;
  __sincosf(pos * inv, &s, &c);
  const float* src = Qf + (size_t)q * HIDDEN + head * HD;
  float x0 = src[i], x1 = src[i + 32];
  __bf16* dst = Qb + ((size_t)head * QLEN + q) * HD;
  dst[i]      = f2bf((x0 * c - x1 * s) * 0.125f);   // fold 1/sqrt(64)
  dst[i + 32] = f2bf((x1 * c + x0 * s) * 0.125f);
}

__global__ void rope_k_kernel(const float* __restrict__ Kf, const int* __restrict__ pid,
                              __bf16* __restrict__ Kb) {
  int t = blockIdx.x * blockDim.x + threadIdx.x; // QLEN*8*32
  int i = t & 31, kvh = (t >> 5) & 7, q = t >> 8;
  float pos = (float)pid[q];
  float inv = __expf(-(float)(2 * i) / (float)HD * __logf(10000.0f));
  float s, c;
  __sincosf(pos * inv, &s, &c);
  const float* src = Kf + (size_t)q * KVDIM + kvh * HD;
  float x0 = src[i], x1 = src[i + 32];
  __bf16* dst = Kb + ((size_t)kvh * QLEN + q) * HD;
  dst[i]      = f2bf(x0 * c - x1 * s);
  dst[i + 32] = f2bf(x1 * c + x0 * s);
}

__global__ void cvt_v_kernel(const float* __restrict__ Vf, __bf16* __restrict__ Vb) {
  int t = blockIdx.x * blockDim.x + threadIdx.x; // QLEN*512
  int q = t >> 9, cidx = t & 511;
  int kvh = cidx >> 6, d = cidx & 63;
  Vb[((size_t)kvh * QLEN + q) * HD + d] = f2bf(Vf[(size_t)q * KVDIM + cidx]);
}

// ---------------- flash attention ----------------
// grid (QLEN/64, 32 heads); block 128 = 4 waves; each wave = 16 q rows
__global__ __launch_bounds__(128) void attn_kernel(const __bf16* __restrict__ Qb,
                                                   const __bf16* __restrict__ Kb,
                                                   const __bf16* __restrict__ Vb,
                                                   const float* __restrict__ maskbuf,
                                                   __bf16* __restrict__ AOb) {
  __shared__ __align__(16) __bf16 Ks[32 * 72];   // [kv][d]
  __shared__ __align__(16) __bf16 Vt[64 * 40];   // [d][kv] (transposed)
  __shared__ __align__(16) __bf16 Ps[4][16 * 40];

  int t = threadIdx.x, lane = t & 31, w = t >> 5;
  int qbase = blockIdx.x * 64, head = blockIdx.y, kvh = head >> 2;
  int wq = qbase + w * 16;
  int ml = lane & 15, h = lane >> 4;

  v16bf qf[2];
#pragma unroll
  for (int dc = 0; dc < 2; ++dc) {
    const __bf16* p = Qb + ((size_t)head * QLEN + wq + ml) * HD + dc * 32 + h * 8;
    v8bf lo = *(const v8bf*)p, hi = *(const v8bf*)(p + 16);
#pragma unroll
    for (int e = 0; e < 8; ++e) { qf[dc][e] = lo[e]; qf[dc][e + 8] = hi[e]; }
  }

  v8f O[4];
  float m_r[8], l_r[8];
#pragma unroll
  for (int dt = 0; dt < 4; ++dt) O[dt] = v8f{0, 0, 0, 0, 0, 0, 0, 0};
#pragma unroll
  for (int r = 0; r < 8; ++r) { m_r[r] = -1e30f; l_r[r] = 0.f; }

  int nkv = qbase / 32 + 2; // causal: last kv block covers qbase+63
  for (int j = 0; j < nkv; ++j) {
    int kvb = j * 32;
    { // K tile: async global->LDS; V tile: manual transpose
      int kv = t >> 2, ds = (t & 3) * 16;
      const __bf16* kp = Kb + ((size_t)kvh * QLEN + kvb + kv) * HD + ds;
      async_b128(lds_off(&Ks[kv * 72 + ds]), kp);
      async_b128(lds_off(&Ks[kv * 72 + ds + 8]), kp + 8);
      const __bf16* vp = Vb + ((size_t)kvh * QLEN + kvb + kv) * HD + ds;
      v8bf v0 = *(const v8bf*)vp, v1 = *(const v8bf*)(vp + 8);
#pragma unroll
      for (int e = 0; e < 8; ++e) {
        Vt[(ds + e) * 40 + kv]     = v0[e];
        Vt[(ds + 8 + e) * 40 + kv] = v1[e];
      }
      wait_async<0>();
    }
    __syncthreads();

    v8f s0 = v8f{0, 0, 0, 0, 0, 0, 0, 0};
    v8f s1 = v8f{0, 0, 0, 0, 0, 0, 0, 0};
#pragma unroll
    for (int dc = 0; dc < 2; ++dc) {
      v16bf b0, b1;
      const __bf16* p0 = &Ks[ml * 72 + dc * 32 + h * 8];
      const __bf16* p1 = &Ks[(16 + ml) * 72 + dc * 32 + h * 8];
      v8bf a0 = *(const v8bf*)p0, a1 = *(const v8bf*)(p0 + 16);
      v8bf c0 = *(const v8bf*)p1, c1 = *(const v8bf*)(p1 + 16);
#pragma unroll
      for (int e = 0; e < 8; ++e) {
        b0[e] = a0[e]; b0[e + 8] = a1[e];
        b1[e] = c0[e]; b1[e + 8] = c1[e];
      }
      s0 = __builtin_amdgcn_wmma_f32_16x16x32_bf16(false, qf[dc], false, b0, (short)0, s0, false, false);
      s1 = __builtin_amdgcn_wmma_f32_16x16x32_bf16(false, qf[dc], false, b1, (short)0, s1, false, false);
    }

    float alpha[8];
#pragma unroll
    for (int r = 0; r < 8; ++r) {
      int qrow = wq + r + (h << 3);
      if (kvb + ml > qrow)      s0[r] = -1e9f;
      if (kvb + 16 + ml > qrow) s1[r] = -1e9f;
      float rm = fmaxf(s0[r], s1[r]);
      for (int x = 8; x >= 1; x >>= 1) rm = fmaxf(rm, __shfl_xor(rm, x, 32));
      float mn = fmaxf(m_r[r], rm);
      alpha[r] = __expf(m_r[r] - mn);
      float p0 = __expf(s0[r] - mn), p1 = __expf(s1[r] - mn);
      float rs = p0 + p1;
      for (int x = 8; x >= 1; x >>= 1) rs += __shfl_xor(rs, x, 32);
      l_r[r] = l_r[r] * alpha[r] + rs;
      m_r[r] = mn;
      int prow = r + (h << 3);
      Ps[w][prow * 40 + ml]      = f2bf(p0);
      Ps[w][prow * 40 + 16 + ml] = f2bf(p1);
    }

    v16bf pf;
    {
      const __bf16* p = &Ps[w][ml * 40 + h * 8];
      v8bf lo = *(const v8bf*)p, hi = *(const v8bf*)(p + 16);
#pragma unroll
      for (int e = 0; e < 8; ++e) { pf[e] = lo[e]; pf[e + 8] = hi[e]; }
    }
#pragma unroll
    for (int dt = 0; dt < 4; ++dt) {
#pragma unroll
      for (int r = 0; r < 8; ++r) O[dt][r] *= alpha[r];
      v16bf bfr;
      const __bf16* p = &Vt[(dt * 16 + ml) * 40 + h * 8];
      v8bf lo = *(const v8bf*)p, hi = *(const v8bf*)(p + 16);
#pragma unroll
      for (int e = 0; e < 8; ++e) { bfr[e] = lo[e]; bfr[e + 8] = hi[e]; }
      O[dt] = __builtin_amdgcn_wmma_f32_16x16x32_bf16(false, pf, false, bfr, (short)0, O[dt], false, false);
    }
    __syncthreads();
  }

#pragma unroll
  for (int dt = 0; dt < 4; ++dt)
#pragma unroll
    for (int r = 0; r < 8; ++r) {
      int qrow = wq + r + (h << 3);
      float mv = maskbuf[(size_t)head * QLEN + qrow];
      float val = O[dt][r] / l_r[r] * mv;
      AOb[(size_t)qrow * HIDDEN + head * HD + dt * 16 + ml] = f2bf(val);
    }
}

// ---------------- host launch ----------------
extern "C" void kernel_launch(void* const* d_in, const int* in_sizes, int n_in,
                              void* d_out, int out_size, void* d_ws, size_t ws_size,
                              hipStream_t stream) {
  const float* X     = (const float*)d_in[0];
  const float* Wq    = (const float*)d_in[1];
  const float* Wk    = (const float*)d_in[2];
  const float* Wv    = (const float*)d_in[3];
  const float* Wo    = (const float*)d_in[4];
  const float* Wpred = (const float*)d_in[5];
  const int*   pid   = (const int*)d_in[6];

  char* ws = (char*)d_ws;
  size_t off = 0;
  auto alloc = [&](size_t bytes) -> void* {
    void* p = ws + off;
    off += (bytes + 255) & ~size_t(255);
    return p;
  };
  __bf16* Xb   = (__bf16*)alloc((size_t)QLEN * HIDDEN * 2);
  __bf16* Wqb  = (__bf16*)alloc((size_t)HIDDEN * HIDDEN * 2);
  __bf16* Wkb  = (__bf16*)alloc((size_t)KVDIM * HIDDEN * 2);
  __bf16* Wvb  = (__bf16*)alloc((size_t)KVDIM * HIDDEN * 2);
  __bf16* Wob  = (__bf16*)alloc((size_t)HIDDEN * HIDDEN * 2);
  float*  Qf   = (float*)alloc((size_t)QLEN * HIDDEN * 4);
  float*  Kf   = (float*)alloc((size_t)QLEN * KVDIM * 4);
  float*  Vf   = (float*)alloc((size_t)QLEN * KVDIM * 4);
  __bf16* Qb   = (__bf16*)alloc((size_t)QLEN * HIDDEN * 2);
  __bf16* Kb   = (__bf16*)alloc((size_t)QLEN * KVDIM * 2);
  __bf16* Vb   = (__bf16*)alloc((size_t)QLEN * KVDIM * 2);
  __bf16* AOb  = (__bf16*)alloc((size_t)QLEN * HIDDEN * 2);
  float*  mask = (float*)alloc((size_t)NHEADS * QLEN * 4);

  const int nHH = HIDDEN * HIDDEN, nKH = KVDIM * HIDDEN;
  cvt_bf16_kernel<<<nHH / 256, 256, 0, stream>>>(X, Xb, nHH);
  cvt_bf16_kernel<<<nHH / 256, 256, 0, stream>>>(Wq, Wqb, nHH);
  cvt_bf16_kernel<<<nKH / 256, 256, 0, stream>>>(Wk, Wkb, nKH);
  cvt_bf16_kernel<<<nKH / 256, 256, 0, stream>>>(Wv, Wvb, nKH);
  cvt_bf16_kernel<<<nHH / 256, 256, 0, stream>>>(Wo, Wob, nHH);

  headmask_kernel<<<QLEN / 8, 256, 0, stream>>>(X, Wpred, mask);

  gemm_bf16_kernel<<<dim3(HIDDEN / 128, QLEN / 128), 256, 0, stream>>>(Xb, Wqb, Qf, QLEN, HIDDEN, HIDDEN);
  gemm_bf16_kernel<<<dim3(KVDIM / 128, QLEN / 128), 256, 0, stream>>>(Xb, Wkb, Kf, QLEN, KVDIM, HIDDEN);
  gemm_bf16_kernel<<<dim3(KVDIM / 128, QLEN / 128), 256, 0, stream>>>(Xb, Wvb, Vf, QLEN, KVDIM, HIDDEN);

  rope_q_kernel<<<(QLEN * NHEADS * 32) / 256, 256, 0, stream>>>(Qf, pid, Qb);
  rope_k_kernel<<<(QLEN * NKV * 32) / 256, 256, 0, stream>>>(Kf, pid, Kb);
  cvt_v_kernel<<<(QLEN * KVDIM) / 256, 256, 0, stream>>>(Vf, Vb);

  attn_kernel<<<dim3(QLEN / 64, NHEADS), 128, 0, stream>>>(Qb, Kb, Vb, mask, AOb);

  gemm_bf16_kernel<<<dim3(HIDDEN / 128, QLEN / 128), 256, 0, stream>>>(AOb, Wob, (float*)d_out, QLEN, HIDDEN, HIDDEN);
}
